// PatchLossOptimizedV2_43052752175787
// MI455X (gfx1250) — matrix operand order repaired and checked by reading, not compiled, not run on hardware
//
#include <hip/hip_runtime.h>
#include <hip/hip_bf16.h>

// Problem constants (match reference)
constexpr int B = 8, T = 2048, D = 128, K = 2;
constexpr int TILES_PER_B = (T / 16) * (T / 16);       // 16384
constexpr int NTILES = B * TILES_PER_B;                // 131072
constexpr int WAVES_PER_BLOCK = 8;                     // 256 threads, wave32
constexpr int NBLOCKS_TILE = NTILES / WAVES_PER_BLOCK; // 16384

typedef __attribute__((ext_vector_type(16))) _Float16 v16h;
typedef __attribute__((ext_vector_type(8)))  _Float16 v8h;
typedef __attribute__((ext_vector_type(8)))  float    v8f;
typedef __attribute__((ext_vector_type(2)))  float    v2f;

// ---------------------------------------------------------------------------
// Kernel 1: z2[row] = sum_d z[row,d]^2 ; split z into f16 hi/lo (z = hi + lo)
// One wave per row of 128 elements; 8 waves per block.
// ---------------------------------------------------------------------------
__global__ void __launch_bounds__(256)
prep_kernel(const float* __restrict__ z, _Float16* __restrict__ zh,
            _Float16* __restrict__ zl, float* __restrict__ z2) {
  const int wave = threadIdx.x >> 5;
  const int lane = threadIdx.x & 31;
  const int row  = blockIdx.x * WAVES_PER_BLOCK + wave;  // [0, B*T)
  const float*    zr  = z  + (size_t)row * D;
  _Float16*       zhr = zh + (size_t)row * D;
  _Float16*       zlr = zl + (size_t)row * D;
  float s = 0.0f;
#pragma unroll
  for (int i = 0; i < 4; ++i) {
    const int idx = lane + i * 32;
    const float x = zr[idx];
    const _Float16 h = (_Float16)x;
    const _Float16 l = (_Float16)(x - (float)h);
    zhr[idx] = h;
    zlr[idx] = l;
    s += x * x;
  }
#pragma unroll
  for (int off = 16; off; off >>= 1) s += __shfl_xor(s, off, 32);
  if (lane == 0) z2[row] = s;
}

// ---------------------------------------------------------------------------
// Kernel 2: one wave32 per 16x16 (i,j) tile.
//   - Issue the 8 NT gt_dT loads (the HBM-roofline stream) FIRST so they are
//     in flight during fragment loads + WMMAs.
//   - G_tile = Z_I * Z_J^T via v_wmma_f32_16x16x32_f16 (hi/lo compensation)
//   - p = sum_{m,n} w_mn * (z2_i[m] + z2_j[n] - 2*G[m,n]),
//     w_mn = exp(-(g0^2/(2s0^2) + g1^2/(2s1^2)))
// ---------------------------------------------------------------------------
__global__ void __launch_bounds__(256)
tile_kernel(const float* __restrict__ gt, const float* __restrict__ sigma,
            const _Float16* __restrict__ zh, const _Float16* __restrict__ zl,
            const float* __restrict__ z2, float* __restrict__ partials) {
  __shared__ float red[WAVES_PER_BLOCK];

  const int wave = threadIdx.x >> 5;
  const int lane = threadIdx.x & 31;
  const int tile = blockIdx.x * WAVES_PER_BLOCK + wave;

  const int b    = tile >> 14;          // tiles per batch = 2^14
  const int rem  = tile & (TILES_PER_B - 1);
  const int ti   = rem >> 7;            // 128 tiles per row-dim
  const int tj   = rem & 127;
  const int row0 = ti << 4;
  const int col0 = tj << 4;

  const int n  = lane & 15;             // M for A-frag, N for B/D-frag
  const int hb = lane >> 4;             // half-wave selector

  // ---- start streaming the gt_dT tile (non-temporal, use-once) ----
  const float* gB = gt + (((size_t)b * T + row0) * T + (col0 + n)) * K;
  v2f g[8];
#pragma unroll
  for (int r = 0; r < 8; ++r) {
    const int m = r + 8 * hb;  // D-matrix layout: VGPR r -> row r (+8 hi half)
    g[r] = __builtin_nontemporal_load((const v2f*)(gB + (size_t)m * T * K));
  }

  // ---- Gram tile via WMMA while gt loads are in flight ----
  const _Float16* zhB = zh + (size_t)b * T * D;
  const _Float16* zlB = zl + (size_t)b * T * D;
  const _Float16* ah  = zhB + (size_t)(row0 + n) * D;  // A row (M = lane&15)
  const _Float16* al  = zlB + (size_t)(row0 + n) * D;
  const _Float16* bh  = zhB + (size_t)(col0 + n) * D;  // B col (N = lane&15)
  const _Float16* bl  = zlB + (size_t)(col0 + n) * D;

  v8f acc = {};
#pragma unroll
  for (int kb = 0; kb < 4; ++kb) {
    const int k0 = kb * 32;
    // A 16x32 f16, contiguous k order used here (same as B): lanes0-15 K=0..15, lanes16-31 K=16..31
    union { v16h v; v8h h[2]; } Ah, Al, Bh, Bl;
    Ah.v = *(const v16h*)(ah + k0 + 16 * hb);
    Al.v = *(const v16h*)(al + k0 + 16 * hb);
    // B 32x16 f16, same contiguous k order as A: lanes0-15 K=0..15, lanes16-31 K=16..31, N=lane&15
    Bh.v = *(const v16h*)(bh + k0 + 16 * hb);
    Bl.v = *(const v16h*)(bl + k0 + 16 * hb);
    acc = __builtin_amdgcn_wmma_f32_16x16x32_f16(false, Ah.v, false, Bh.v,
                                                 (short)0, acc, false, false);
    acc = __builtin_amdgcn_wmma_f32_16x16x32_f16(false, Ah.v, false, Bl.v,
                                                 (short)0, acc, false, false);
    acc = __builtin_amdgcn_wmma_f32_16x16x32_f16(false, Al.v, false, Bh.v,
                                                 (short)0, acc, false, false);
  }

  const float s0 = sigma[0], s1 = sigma[1];
  const float inv0 = 0.5f / (s0 * s0);
  const float inv1 = 0.5f / (s1 * s1);

  const float* z2b = z2 + (size_t)b * T;
  const float  z2n = z2b[col0 + n];

  float p = 0.0f;
#pragma unroll
  for (int r = 0; r < 8; ++r) {
    const int m = r + 8 * hb;
    const float kern = g[r].x * g[r].x * inv0 + g[r].y * g[r].y * inv1;
    const float w = __expf(-kern);
    p += w * (z2b[row0 + m] + z2n - 2.0f * acc[r]);
  }

#pragma unroll
  for (int off = 16; off; off >>= 1) p += __shfl_xor(p, off, 32);
  if (lane == 0) red[wave] = p;
  __syncthreads();
  if (threadIdx.x == 0) {
    float t = 0.0f;
#pragma unroll
    for (int i = 0; i < WAVES_PER_BLOCK; ++i) t += red[i];
    partials[blockIdx.x] = t;
  }
}

// ---------------------------------------------------------------------------
// Kernel 3: deterministic final reduction of 16384 partials -> scalar loss
// ---------------------------------------------------------------------------
__global__ void __launch_bounds__(256)
reduce_kernel(const float* __restrict__ partials, float* __restrict__ out) {
  __shared__ float red[256];
  float s = 0.0f;
  for (int i = threadIdx.x; i < NBLOCKS_TILE; i += 256) s += partials[i];
  red[threadIdx.x] = s;
  __syncthreads();
  for (int off = 128; off; off >>= 1) {
    if (threadIdx.x < off) red[threadIdx.x] += red[threadIdx.x + off];
    __syncthreads();
  }
  if (threadIdx.x == 0)
    out[0] = red[0] * (1.0f / ((float)B * (float)T * (float)T));
}

// ---------------------------------------------------------------------------
extern "C" void kernel_launch(void* const* d_in, const int* in_sizes, int n_in,
                              void* d_out, int out_size, void* d_ws,
                              size_t ws_size, hipStream_t stream) {
  const float* z     = (const float*)d_in[0];  // [B,T,D] f32
  const float* gt    = (const float*)d_in[1];  // [B,T,T,K] f32
  const float* sigma = (const float*)d_in[2];  // [K] f32
  float*       out   = (float*)d_out;          // scalar f32

  // Workspace layout
  const size_t ZBYTES  = (size_t)B * T * D * sizeof(_Float16);  // 4 MB
  const size_t Z2BYTES = (size_t)B * T * sizeof(float);         // 64 KB
  char* ws = (char*)d_ws;
  _Float16* zh       = (_Float16*)ws;
  _Float16* zl       = (_Float16*)(ws + ZBYTES);
  float*    z2       = (float*)(ws + 2 * ZBYTES);
  float*    partials = (float*)(ws + 2 * ZBYTES + Z2BYTES);

  prep_kernel<<<(B * T) / WAVES_PER_BLOCK, 256, 0, stream>>>(z, zh, zl, z2);
  tile_kernel<<<NBLOCKS_TILE, 256, 0, stream>>>(gt, sigma, zh, zl, z2,
                                                partials);
  reduce_kernel<<<1, 256, 0, stream>>>(partials, out);
}
